// GIN_49014166782120
// MI455X (gfx1250) — compile-verified, matrix operand order checked by reading, not run on hardware
//
#include <hip/hip_runtime.h>

// GIN forward, MI455X (gfx1250, wave32).
// GEMMs via v_wmma_f32_16x16x32_bf16 (bf16 in, f32 accum); scatter via f32 L2 atomics.
// f32->bf16 via native casts so the backend emits v_cvt_pk_bf16_f32 (not scalar bit math).

#define NNODES 50000
#define NEDGES 800000
#define DH 128
#define DOUT 40
#define DOUTP 48
#define BN_EPS 1e-5f

typedef __attribute__((ext_vector_type(16))) __bf16 v16bf;
typedef __attribute__((ext_vector_type(8)))  float  v8f;

// ---- weight prep: W[K=128][Nout] f32 -> WT[NoutPad][128] bf16 (transposed, zero-padded)
__global__ __launch_bounds__(256) void wprep(const float* __restrict__ W,
                                             __bf16* __restrict__ WT,
                                             int Nout, int NoutPad) {
  int i = blockIdx.x * 256 + threadIdx.x;
  if (i >= NoutPad * DH) return;
  int n = i >> 7, k = i & 127;
  float v = (n < Nout) ? W[k * Nout + n] : 0.f;
  WT[i] = (__bf16)v;
}

__global__ void biaspad(const float* __restrict__ b, float* __restrict__ bp, int n, int npad) {
  int i = threadIdx.x;
  if (i < npad) bp[i] = (i < n) ? b[i] : 0.f;
}

// ---- agg = src (vectorized copy), then edge kernel atomically adds neighbors
__global__ __launch_bounds__(256) void copy4(const float4* __restrict__ s,
                                             float4* __restrict__ d, int n4) {
  int i = blockIdx.x * 256 + threadIdx.x;
  if (i < n4) d[i] = s[i];
}

__global__ __launch_bounds__(256) void edge_scatter(const long long* __restrict__ ei,
                                                    const float* __restrict__ x,
                                                    float* __restrict__ agg, int E) {
  int gid = blockIdx.x * 256 + threadIdx.x;
  int e = gid >> 5;
  if (e >= E) return;
  int c = (gid & 31) * 4;
  long long s = ei[e];        // row 0: src
  long long d = ei[E + e];    // row 1: dst
  float4 v = *(const float4*)&x[(size_t)s * DH + c];
  float* p = &agg[(size_t)d * DH + c];
  atomicAdd(p + 0, v.x);
  atomicAdd(p + 1, v.y);
  atomicAdd(p + 2, v.z);
  atomicAdd(p + 3, v.w);
}

// ---- WMMA GEMM: out[M x ldo] = epilogue( pre(A)[M x 128] @ WT^T + bias )
// pre(A): optional per-column BN scale/shift + ReLU, fused into bf16 convert.
// Block: 256 threads = 8 waves; wave -> 16 rows x (NT*16) cols; K=128 in 4 steps of 32.
template<int NT, bool PRE_BN, bool POST_RELU>
__global__ __launch_bounds__(256) void gemm_wmma(const float* __restrict__ A,
                                                 const __bf16* __restrict__ WT,
                                                 const float* __restrict__ bias,
                                                 const float* __restrict__ scale,
                                                 const float* __restrict__ shift,
                                                 float* __restrict__ out,
                                                 int M, int ldo) {
  __shared__ __bf16 lW[NT * 16 * DH];
  const int tid = threadIdx.x;
  for (int i = tid * 8; i < NT * 16 * DH; i += 256 * 8)
    *(uint4*)&lW[i] = *(const uint4*)&WT[i];
  __syncthreads();

  const int wave  = tid >> 5;
  const int lane  = tid & 31;
  const int mbase = blockIdx.x * 128 + wave * 16;
  if (mbase >= M) return;                  // uniform per wave; M % 16 == 0
  const int   loff = (lane & 16) ? 8 : 0;  // ISA 16-bit A-frag half-wave K offset
  const int   row  = mbase + (lane & 15);
  const float* arow = A + (size_t)row * DH + loff;

  v8f acc[NT] = {};

#pragma unroll
  for (int ks = 0; ks < 4; ++ks) {
    const int kb = ks * 32;
    float4 a0 = *(const float4*)(arow + kb);
    float4 a1 = *(const float4*)(arow + kb + 4);
    float4 a2 = *(const float4*)(arow + kb + 16);
    float4 a3 = *(const float4*)(arow + kb + 20);
    if (PRE_BN) {
      const int k0 = kb + loff;
      float4 s0 = *(const float4*)(scale + k0);
      float4 s1 = *(const float4*)(scale + k0 + 4);
      float4 s2 = *(const float4*)(scale + k0 + 16);
      float4 s3 = *(const float4*)(scale + k0 + 20);
      float4 h0 = *(const float4*)(shift + k0);
      float4 h1 = *(const float4*)(shift + k0 + 4);
      float4 h2 = *(const float4*)(shift + k0 + 16);
      float4 h3 = *(const float4*)(shift + k0 + 20);
      a0.x = fmaxf(fmaf(a0.x, s0.x, h0.x), 0.f);
      a0.y = fmaxf(fmaf(a0.y, s0.y, h0.y), 0.f);
      a0.z = fmaxf(fmaf(a0.z, s0.z, h0.z), 0.f);
      a0.w = fmaxf(fmaf(a0.w, s0.w, h0.w), 0.f);
      a1.x = fmaxf(fmaf(a1.x, s1.x, h1.x), 0.f);
      a1.y = fmaxf(fmaf(a1.y, s1.y, h1.y), 0.f);
      a1.z = fmaxf(fmaf(a1.z, s1.z, h1.z), 0.f);
      a1.w = fmaxf(fmaf(a1.w, s1.w, h1.w), 0.f);
      a2.x = fmaxf(fmaf(a2.x, s2.x, h2.x), 0.f);
      a2.y = fmaxf(fmaf(a2.y, s2.y, h2.y), 0.f);
      a2.z = fmaxf(fmaf(a2.z, s2.z, h2.z), 0.f);
      a2.w = fmaxf(fmaf(a2.w, s2.w, h2.w), 0.f);
      a3.x = fmaxf(fmaf(a3.x, s3.x, h3.x), 0.f);
      a3.y = fmaxf(fmaf(a3.y, s3.y, h3.y), 0.f);
      a3.z = fmaxf(fmaf(a3.z, s3.z, h3.z), 0.f);
      a3.w = fmaxf(fmaf(a3.w, s3.w, h3.w), 0.f);
    }
    // native casts -> v_cvt_pk_bf16_f32 (RNE), one op per bf16 pair
    v16bf af;
    af[0]  = (__bf16)a0.x; af[1]  = (__bf16)a0.y;
    af[2]  = (__bf16)a0.z; af[3]  = (__bf16)a0.w;
    af[4]  = (__bf16)a1.x; af[5]  = (__bf16)a1.y;
    af[6]  = (__bf16)a1.z; af[7]  = (__bf16)a1.w;
    af[8]  = (__bf16)a2.x; af[9]  = (__bf16)a2.y;
    af[10] = (__bf16)a2.z; af[11] = (__bf16)a2.w;
    af[12] = (__bf16)a3.x; af[13] = (__bf16)a3.y;
    af[14] = (__bf16)a3.z; af[15] = (__bf16)a3.w;
#pragma unroll
    for (int nt = 0; nt < NT; ++nt) {
      const __bf16* bp = &lW[(nt * 16 + (lane & 15)) * DH + kb + loff];
      v16bf bf;
      uint4* bu = (uint4*)&bf;
      bu[0] = *(const uint4*)bp;         // K pairs (k0,k1)..(k6,k7)
      bu[1] = *(const uint4*)(bp + 16);  // K pairs (k16,k17)..(k22,k23)
      acc[nt] = __builtin_amdgcn_wmma_f32_16x16x32_bf16(
          false, af, false, bf, (short)0, acc[nt], false, false);
    }
  }

#pragma unroll
  for (int nt = 0; nt < NT; ++nt) {
    const int n = nt * 16 + (lane & 15);
    const float b = bias[n];
    const int r0 = (lane & 16) ? 8 : 0;
#pragma unroll
    for (int r = 0; r < 8; ++r) {
      float v = acc[nt][r] + b;
      if (POST_RELU) v = fmaxf(v, 0.f);
      out[(size_t)(mbase + r0 + r) * ldo + n] = v;
    }
  }
}

// ---- BN column statistics: sums[c] += sum(t[:,c]), sumsq[c] += sum(t[:,c]^2)
__global__ __launch_bounds__(256) void bn_stats(const float* __restrict__ t,
                                                float* __restrict__ sums,
                                                float* __restrict__ sumsq, int M) {
  const int c  = threadIdx.x & 127;
  const int rg = threadIdx.x >> 7;  // 0 or 1
  const int row0 = blockIdx.x * 128;
  const int rend = min(row0 + 128, M);
  float s = 0.f, q = 0.f;
  for (int r = row0 + rg; r < rend; r += 2) {
    float v = t[(size_t)r * DH + c];
    s += v;
    q += v * v;
  }
  __shared__ float ls[256], lq[256];
  ls[threadIdx.x] = s;
  lq[threadIdx.x] = q;
  __syncthreads();
  if (rg == 0) {
    atomicAdd(&sums[c], s + ls[c + 128]);
    atomicAdd(&sumsq[c], q + lq[c + 128]);
  }
}

__global__ void zero_f32(float* p, int n) {
  int i = blockIdx.x * 256 + threadIdx.x;
  if (i < n) p[i] = 0.f;
}

__global__ void bn_finalize(const float* __restrict__ sums, const float* __restrict__ sumsq,
                            const float* __restrict__ gamma, const float* __restrict__ beta,
                            float* __restrict__ scale, float* __restrict__ shift, float invN) {
  int c = threadIdx.x;  // 128 threads
  float mu  = sums[c] * invN;
  float var = sumsq[c] * invN - mu * mu;   // biased variance (matches jnp.var)
  float sc  = gamma[c] * rsqrtf(var + BN_EPS);
  scale[c] = sc;
  shift[c] = beta[c] - mu * sc;
}

// ---- log-softmax over 40 cols; one wave32 per row
__global__ __launch_bounds__(256) void logsoftmax(const float* __restrict__ logits,
                                                  float* __restrict__ out, int M) {
  const int wave = threadIdx.x >> 5;
  const int lane = threadIdx.x & 31;
  const int row = blockIdx.x * 8 + wave;
  if (row >= M) return;
  const float* lr = logits + (size_t)row * DOUTP;
  float v0 = lr[lane];                                        // cols 0..31
  float v1 = (lane + 32 < DOUT) ? lr[lane + 32] : -3.0e38f;   // cols 32..39
  float m = fmaxf(v0, v1);
  for (int off = 16; off > 0; off >>= 1) m = fmaxf(m, __shfl_xor(m, off, 32));
  float e = __expf(v0 - m) + ((lane + 32 < DOUT) ? __expf(v1 - m) : 0.f);
  for (int off = 16; off > 0; off >>= 1) e += __shfl_xor(e, off, 32);
  float lse = __logf(e) + m;
  out[(size_t)row * DOUT + lane] = v0 - lse;
  if (lane + 32 < DOUT) out[(size_t)row * DOUT + lane + 32] = v1 - lse;
}

extern "C" void kernel_launch(void* const* d_in, const int* in_sizes, int n_in,
                              void* d_out, int out_size, void* d_ws, size_t ws_size,
                              hipStream_t stream) {
  const float*     x   = (const float*)d_in[0];
  const long long* ei  = (const long long*)d_in[1];  // int64 per reference
  const float* W1a = (const float*)d_in[2];
  const float* b1a = (const float*)d_in[3];
  const float* g1  = (const float*)d_in[4];
  const float* be1 = (const float*)d_in[5];
  const float* W2a = (const float*)d_in[6];
  const float* b2a = (const float*)d_in[7];
  const float* W1b = (const float*)d_in[8];
  const float* b1b = (const float*)d_in[9];
  const float* g2  = (const float*)d_in[10];
  const float* be2 = (const float*)d_in[11];
  const float* W2b = (const float*)d_in[12];
  const float* b2b = (const float*)d_in[13];
  const float* Wl1 = (const float*)d_in[14];
  const float* bl1 = (const float*)d_in[15];
  const float* Wl2 = (const float*)d_in[16];
  const float* bl2 = (const float*)d_in[17];
  float* outp = (float*)d_out;

  // workspace layout
  const size_t nb = (size_t)NNODES * DH;
  float* buf0 = (float*)d_ws;       // agg
  float* buf1 = buf0 + nb;          // pre-BN / logits
  float* buf2 = buf1 + nb;          // activations
  float* sums   = buf2 + nb;        // 128
  float* sumsq  = sums + 128;       // 128
  float* scalev = sumsq + 128;      // 128
  float* shiftv = scalev + 128;     // 128
  float* blpad  = shiftv + 128;     // 48 (padded to 64)
  __bf16* wtW1a = (__bf16*)(blpad + 64);
  __bf16* wtW2a = wtW1a + DH * DH;
  __bf16* wtW1b = wtW2a + DH * DH;
  __bf16* wtW2b = wtW1b + DH * DH;
  __bf16* wtWl1 = wtW2b + DH * DH;
  __bf16* wtWl2 = wtWl1 + DH * DH;  // DOUTP*DH

  const int wgrid  = (DH * DH + 255) / 256;
  const int wgrid2 = (DOUTP * DH + 255) / 256;
  wprep<<<wgrid,  256, 0, stream>>>(W1a, wtW1a, DH, DH);
  wprep<<<wgrid,  256, 0, stream>>>(W2a, wtW2a, DH, DH);
  wprep<<<wgrid,  256, 0, stream>>>(W1b, wtW1b, DH, DH);
  wprep<<<wgrid,  256, 0, stream>>>(W2b, wtW2b, DH, DH);
  wprep<<<wgrid,  256, 0, stream>>>(Wl1, wtWl1, DH, DH);
  wprep<<<wgrid2, 256, 0, stream>>>(Wl2, wtWl2, DOUT, DOUTP);
  biaspad<<<1, 64, 0, stream>>>(bl2, blpad, DOUT, DOUTP);

  const int cgrid = (int)((nb / 4 + 255) / 256);
  const int egrid = (NEDGES * 32 + 255) / 256;
  const int ggrid = (NNODES + 127) / 128;
  const int sgrid = (NNODES + 127) / 128;
  const int lgrid = (NNODES + 7) / 8;
  const float invN = 1.0f / (float)NNODES;

  // ---- conv1 ----
  copy4<<<cgrid, 256, 0, stream>>>((const float4*)x, (float4*)buf0, (int)(nb / 4));
  edge_scatter<<<egrid, 256, 0, stream>>>(ei, x, buf0, NEDGES);
  gemm_wmma<8, false, false><<<ggrid, 256, 0, stream>>>(buf0, wtW1a, b1a, nullptr, nullptr,
                                                        buf1, NNODES, DH);
  zero_f32<<<1, 256, 0, stream>>>(sums, 256);  // sums+sumsq contiguous
  bn_stats<<<sgrid, 256, 0, stream>>>(buf1, sums, sumsq, NNODES);
  bn_finalize<<<1, 128, 0, stream>>>(sums, sumsq, g1, be1, scalev, shiftv, invN);
  gemm_wmma<8, true, true><<<ggrid, 256, 0, stream>>>(buf1, wtW2a, b2a, scalev, shiftv,
                                                      buf2, NNODES, DH);  // h1 (outer relu = relu)

  // ---- conv2 ----
  copy4<<<cgrid, 256, 0, stream>>>((const float4*)buf2, (float4*)buf0, (int)(nb / 4));
  edge_scatter<<<egrid, 256, 0, stream>>>(ei, buf2, buf0, NEDGES);
  gemm_wmma<8, false, false><<<ggrid, 256, 0, stream>>>(buf0, wtW1b, b1b, nullptr, nullptr,
                                                        buf1, NNODES, DH);
  zero_f32<<<1, 256, 0, stream>>>(sums, 256);
  bn_stats<<<sgrid, 256, 0, stream>>>(buf1, sums, sumsq, NNODES);
  bn_finalize<<<1, 128, 0, stream>>>(sums, sumsq, g2, be2, scalev, shiftv, invN);
  gemm_wmma<8, true, true><<<ggrid, 256, 0, stream>>>(buf1, wtW2b, b2b, scalev, shiftv,
                                                      buf2, NNODES, DH);  // h2

  // ---- head ----
  gemm_wmma<8, false, true><<<ggrid, 256, 0, stream>>>(buf2, wtWl1, bl1, nullptr, nullptr,
                                                       buf0, NNODES, DH);       // h3
  gemm_wmma<3, false, false><<<ggrid, 256, 0, stream>>>(buf0, wtWl2, blpad, nullptr, nullptr,
                                                        buf1, NNODES, DOUTP);   // logits
  logsoftmax<<<lgrid, 256, 0, stream>>>(buf1, outp, NNODES);
}
